// HawkesLSTMCell_74938589381016
// MI455X (gfx1250) — matrix operand order repaired and checked by reading, not compiled, not run on hardware
//
#include <hip/hip_runtime.h>
#include <hip/hip_bf16.h>
#include <math.h>

// Problem constants (from reference): B=8192, D=512, H=1024, G=7
#define B_   8192
#define D_   512
#define H_   1024
#define G_   7
#define K1_  (D_ + H_)   // 1536

// LDS row padding: BK + 8 elements -> 144B rows (36 DWORD stride, conflict-free)
#define BK_   64
#define LDA_  (BK_ + 8)  // 72

static_assert((K1_ / BK_) % 2 == 0, "gemm1 K-loop must have even trip count for double buffering");
static_assert((H_ / BK_) % 2 == 0, "gemm2 K-loop must have even trip count for double buffering");

typedef __attribute__((ext_vector_type(16))) __bf16 v16bf;
typedef __attribute__((ext_vector_type(8)))  __bf16 v8bf;
typedef __attribute__((ext_vector_type(8)))  float  v8f;

// ---------------------------------------------------------------------------
// Async global->LDS (CDNA5 GLOBAL_LOAD_ASYNC_TO_LDS_B128, tracked by ASYNCcnt).
// Falls back to load+ds_store if the builtin is unavailable on this toolchain.
#if defined(__has_builtin)
#if __has_builtin(__builtin_amdgcn_global_load_async_to_lds_b128)
#define USE_ASYNC_LDS 1
#endif
#endif
#ifndef USE_ASYNC_LDS
#define USE_ASYNC_LDS 0
#endif

#if USE_ASYNC_LDS
// Parameter type per hipcc diagnostic: pointer to 16B int vector, AS(1)/AS(3).
typedef int v4i_ __attribute__((vector_size(16)));
typedef __attribute__((address_space(1))) v4i_* as1_v4i_ptr;
typedef __attribute__((address_space(3))) v4i_* as3_v4i_ptr;
#endif

static __device__ __forceinline__ void copy16_g2l(const __bf16* g, __bf16* l) {
#if USE_ASYNC_LDS
    // Generic LDS address truncates to the LDS byte offset (ISA 10.2), so the
    // low 32 bits give a valid addrspace(3) pointer. Global VA is used as-is.
    __builtin_amdgcn_global_load_async_to_lds_b128(
        (as1_v4i_ptr)(uintptr_t)g,
        (as3_v4i_ptr)(unsigned)(uintptr_t)l,
        0, 0);
#else
    *(v8bf*)l = *(const v8bf*)g;
#endif
}

static __device__ __forceinline__ void async_stage_wait() {
#if USE_ASYNC_LDS
    asm volatile("s_wait_asynccnt 0x0" ::: "memory");
#endif
    // Fallback path: __syncthreads() below inserts the DScnt waits.
}

// ---------------------------------------------------------------------------
// Fragment loaders per CDNA5 ISA 7.12.2 VGPR layouts (wave32).
//
// A (16x32 bf16, M x K): lane holds row m (row arg includes lane&15).
//   lanes 0-15 : elems 0..7 = K 0..7,  elems 8..15 = K 16..23
//   lanes 16-31: elems 0..7 = K 8..15, elems 8..15 = K 24..31
static __device__ __forceinline__ v16bf load_a_frag(const __bf16* base, int ldk,
                                                    int row, int kbase, int lane) {
    const int laneHi = lane >> 4;
    const __bf16* p = base + (size_t)row * (size_t)ldk + (size_t)(kbase + (laneHi ? 8 : 0));
    v8bf lo = *(const v8bf*)(p);
    v8bf hi = *(const v8bf*)(p + 16);
    return __builtin_shufflevector(lo, hi, 0,1,2,3,4,5,6,7,8,9,10,11,12,13,14,15);
}

// B (32x16 bf16, K x N): lane holds column n; lanes 0-15 K=0..7,16..23, 16-31 K=8..15,24..31.
// Stored [n][k] row-major -> two contiguous 16-byte loads per lane.
static __device__ __forceinline__ v16bf load_b_frag(const __bf16* base, int ldk,
                                                    int nrow, int kbase, int lane) {
    const int laneHi = lane >> 4;
    const __bf16* p = base + (size_t)nrow * (size_t)ldk + (size_t)(kbase + 8 * laneHi);
    v8bf lo = *(const v8bf*)(p);
    v8bf hi = *(const v8bf*)(p + 16);
    return __builtin_shufflevector(lo, hi, 0,1,2,3,4,5,6,7,8,9,10,11,12,13,14,15);
}

static __device__ __forceinline__ float sigmoidf_(float x) {
    return 1.0f / (1.0f + expf(-x));
}
static __device__ __forceinline__ float softplusf_(float x) {
    return fmaxf(x, 0.0f) + log1pf(expf(-fabsf(x)));  // logaddexp(x, 0)
}

// ---------------------------------------------------------------------------
// Kernel 1: V = bf16(concat(x, h_t))  [B, K1]
__global__ void build_v_kernel(const float* __restrict__ x, const float* __restrict__ h,
                               __bf16* __restrict__ v) {
    size_t i = (size_t)blockIdx.x * blockDim.x + threadIdx.x;
    if (i >= (size_t)B_ * K1_) return;
    int b = (int)(i / K1_);
    int k = (int)(i % K1_);
    float val = (k < D_) ? x[(size_t)b * D_ + k] : h[(size_t)b * H_ + (k - D_)];
    v[i] = (__bf16)val;
}

// Kernel 2: fp32 -> bf16 (weights keep native [g][n][k] layout = B-operand layout)
__global__ void cvt_bf16_kernel(const float* __restrict__ src, __bf16* __restrict__ dst,
                                size_t n) {
    size_t i = (size_t)blockIdx.x * blockDim.x + threadIdx.x;
    if (i >= n) return;
    dst[i] = (__bf16)src[i];
}

// ---------------------------------------------------------------------------
// Kernel 3: h1[g] = tanh(V @ W1[g]^T + b1[g]) -> bf16 [G,B,H]
// Block tile 128(M) x 64(N), BK=64; double-buffered LDS staging via async
// global->LDS copies overlapped with WMMA. 8 waves, 32x32 wave tile each.
__global__ __launch_bounds__(256) void gemm1_tanh_kernel(
    const __bf16* __restrict__ Vbf, const __bf16* __restrict__ W1bf,
    const float* __restrict__ b1, __bf16* __restrict__ h1bf) {
    __shared__ __bf16 As[2][128 * LDA_];   // 2 x 18432 B
    __shared__ __bf16 Bs[2][64 * LDA_];    // 2 x  9216 B

    const int tid  = threadIdx.x;
    const int lane = tid & 31;
    const int wave = tid >> 5;        // 0..7
    const int wm   = wave >> 1;       // 0..3  (M direction)
    const int wn   = wave & 1;        // 0..1  (N direction)

    int blk = blockIdx.x;
    const int nBN = H_ / 64;          // 16
    const int nBM = B_ / 128;         // 64
    const int bn = blk % nBN; blk /= nBN;
    const int bm = blk % nBM; blk /= nBM;
    const int g  = blk;
    if (g >= G_) return;

    const __bf16* Wg = W1bf + (size_t)g * H_ * K1_;
    const int gm0 = bm * 128, gn0 = bn * 64;

    auto stage = [&](int buf, int kk) {
        // A: 128 rows x 64 k = 1024 16B-chunks, 4 per thread
        #pragma unroll
        for (int c = 0; c < 4; ++c) {
            int ch  = tid * 4 + c;
            int row = ch >> 3;                // 8 chunks per row
            int ko  = (ch & 7) * 8;
            copy16_g2l(&Vbf[(size_t)(gm0 + row) * K1_ + kk + ko],
                       &As[buf][row * LDA_ + ko]);
        }
        // B: 64 rows x 64 k = 512 chunks, 2 per thread
        #pragma unroll
        for (int c = 0; c < 2; ++c) {
            int ch  = tid * 2 + c;
            int row = ch >> 3;
            int ko  = (ch & 7) * 8;
            copy16_g2l(&Wg[(size_t)(gn0 + row) * K1_ + kk + ko],
                       &Bs[buf][row * LDA_ + ko]);
        }
    };

    v8f acc[2][2] = {};

    stage(0, 0);
    int cur = 0;
    for (int kk = 0; kk < K1_; kk += BK_) {
        async_stage_wait();
        __syncthreads();
        if (kk + BK_ < K1_) stage(cur ^ 1, kk + BK_);

        #pragma unroll
        for (int ks = 0; ks < BK_; ks += 32) {
            v16bf af[2], bfr[2];
            #pragma unroll
            for (int i = 0; i < 2; ++i)
                af[i] = load_a_frag(As[cur], LDA_, wm * 32 + i * 16 + (lane & 15), ks, lane);
            #pragma unroll
            for (int j = 0; j < 2; ++j)
                bfr[j] = load_b_frag(Bs[cur], LDA_, wn * 32 + j * 16 + (lane & 15), ks, lane);
            #pragma unroll
            for (int i = 0; i < 2; ++i)
                #pragma unroll
                for (int j = 0; j < 2; ++j)
                    acc[i][j] = __builtin_amdgcn_wmma_f32_16x16x32_bf16(
                        false, af[i], false, bfr[j], (short)0, acc[i][j], false, false);
        }
        cur ^= 1;
    }

    // Epilogue: + b1, tanh, store bf16
    const int laneHi = lane >> 4;
    __bf16* outg = h1bf + (size_t)g * B_ * H_;
    #pragma unroll
    for (int j = 0; j < 2; ++j) {
        const int n = gn0 + wn * 32 + j * 16 + (lane & 15);
        const float bias = b1[g * H_ + n];
        #pragma unroll
        for (int i = 0; i < 2; ++i) {
            #pragma unroll
            for (int r = 0; r < 8; ++r) {
                int m = gm0 + wm * 32 + i * 16 + r + 8 * laneHi;
                outg[(size_t)m * H_ + n] = (__bf16)tanhf(acc[i][j][r] + bias);
            }
        }
    }
}

// ---------------------------------------------------------------------------
// Kernel 4: fused second GEMMs + Hawkes-LSTM gating.
// Block tile 64(M) x 32(N); 8 waves each own one 16x16 tile and accumulate
// all 7 gates sequentially (56 acc VGPRs); double-buffered async LDS staging.
__global__ __launch_bounds__(256) void gemm2_gate_kernel(
    const __bf16* __restrict__ h1bf, const __bf16* __restrict__ W2bf,
    const float* __restrict__ b2, const float* __restrict__ c_t,
    const float* __restrict__ c_target, float* __restrict__ out) {
    __shared__ __bf16 As[2][64 * LDA_];    // 2 x 9216 B
    __shared__ __bf16 Bs[2][32 * LDA_];    // 2 x 4608 B

    const int tid  = threadIdx.x;
    const int lane = tid & 31;
    const int wave = tid >> 5;        // 0..7
    const int wm   = wave >> 1;       // 0..3
    const int wn   = wave & 1;        // 0..1

    const int nBN = H_ / 32;          // 32
    const int bn = blockIdx.x % nBN;
    const int bm = blockIdx.x / nBN;
    if (bm >= B_ / 64) return;
    const int gm0 = bm * 64, gn0 = bn * 32;

    const int arow = wm * 16 + (lane & 15);   // row within As
    const int brow = wn * 16 + (lane & 15);   // row within Bs
    const int ncol = gn0 + wn * 16 + (lane & 15);

    float gr[G_][8];
    #pragma unroll
    for (int g = 0; g < G_; ++g) {
        const __bf16* Ag = h1bf + (size_t)g * B_ * H_;
        const __bf16* Wg = W2bf + (size_t)g * H_ * H_;

        auto stage = [&](int buf, int kk) {
            // A: 64 x 64 = 512 chunks, 2 per thread
            #pragma unroll
            for (int c = 0; c < 2; ++c) {
                int ch  = tid * 2 + c;
                int row = ch >> 3;
                int ko  = (ch & 7) * 8;
                copy16_g2l(&Ag[(size_t)(gm0 + row) * H_ + kk + ko],
                           &As[buf][row * LDA_ + ko]);
            }
            // B: 32 x 64 = 256 chunks, one per thread
            {
                int row = tid >> 3;
                int ko  = (tid & 7) * 8;
                copy16_g2l(&Wg[(size_t)(gn0 + row) * H_ + kk + ko],
                           &Bs[buf][row * LDA_ + ko]);
            }
        };

        v8f acc = {};
        stage(0, 0);       // safe: buffer 0's last readers were >=1 barrier ago
        int cur = 0;
        for (int kk = 0; kk < H_; kk += BK_) {
            async_stage_wait();
            __syncthreads();
            if (kk + BK_ < H_) stage(cur ^ 1, kk + BK_);
            #pragma unroll
            for (int ks = 0; ks < BK_; ks += 32) {
                v16bf a = load_a_frag(As[cur], LDA_, arow, ks, lane);
                v16bf b = load_b_frag(Bs[cur], LDA_, brow, ks, lane);
                acc = __builtin_amdgcn_wmma_f32_16x16x32_bf16(
                    false, a, false, b, (short)0, acc, false, false);
            }
            cur ^= 1;
        }
        const float bias = b2[g * H_ + ncol];
        #pragma unroll
        for (int r = 0; r < 8; ++r) gr[g][r] = acc[r] + bias;
    }

    // Fused Hawkes-LSTM epilogue
    const int laneHi = lane >> 4;
    const size_t BH = (size_t)B_ * H_;
    #pragma unroll
    for (int r = 0; r < 8; ++r) {
        size_t idx = (size_t)(gm0 + wm * 16 + r + 8 * laneHi) * H_ + ncol;
        float inpt   = sigmoidf_(gr[0][r]);
        float forget = sigmoidf_(gr[1][r]);
        float itgt   = sigmoidf_(gr[2][r]);
        float ftgt   = sigmoidf_(gr[3][r]);
        float outp   = sigmoidf_(gr[4][r]);
        float z      = tanhf(gr[5][r]);
        float decay  = softplusf_(gr[6][r]);
        float ct  = c_t[idx];
        float ctg = c_target[idx];
        out[idx]          = forget * ct  + inpt * z;   // c_i
        out[BH + idx]     = ftgt  * ctg + itgt * z;    // c_target_new
        out[2 * BH + idx] = outp;                      // output gate
        out[3 * BH + idx] = decay;                     // decay
    }
}

// ---------------------------------------------------------------------------
extern "C" void kernel_launch(void* const* d_in, const int* in_sizes, int n_in,
                              void* d_out, int out_size, void* d_ws, size_t ws_size,
                              hipStream_t stream) {
    (void)in_sizes; (void)n_in; (void)out_size; (void)ws_size;
    const float* x        = (const float*)d_in[0];
    const float* h_t      = (const float*)d_in[1];
    const float* c_t      = (const float*)d_in[2];
    const float* c_target = (const float*)d_in[3];
    const float* W1       = (const float*)d_in[4];
    const float* b1       = (const float*)d_in[5];
    const float* W2       = (const float*)d_in[6];
    const float* b2       = (const float*)d_in[7];
    float* out = (float*)d_out;

    const size_t V_ELEMS  = (size_t)B_ * K1_;          // 12.58M
    const size_t W1_ELEMS = (size_t)G_ * H_ * K1_;     // 11.01M
    const size_t W2_ELEMS = (size_t)G_ * H_ * H_;      //  7.34M
    char* ws = (char*)d_ws;
    __bf16* Vbf  = (__bf16*)ws;
    __bf16* W1bf = (__bf16*)(ws + V_ELEMS * 2);
    __bf16* W2bf = (__bf16*)(ws + (V_ELEMS + W1_ELEMS) * 2);
    __bf16* h1bf = (__bf16*)(ws + (V_ELEMS + W1_ELEMS + W2_ELEMS) * 2);

    const int TPB = 256;

    {   // 1) V = bf16(concat(x, h_t))
        dim3 grid((unsigned)((V_ELEMS + TPB - 1) / TPB));
        build_v_kernel<<<grid, TPB, 0, stream>>>(x, h_t, Vbf);
    }
    {   // 2) weights -> bf16
        dim3 g1((unsigned)((W1_ELEMS + TPB - 1) / TPB));
        cvt_bf16_kernel<<<g1, TPB, 0, stream>>>(W1, W1bf, W1_ELEMS);
        dim3 g2((unsigned)((W2_ELEMS + TPB - 1) / TPB));
        cvt_bf16_kernel<<<g2, TPB, 0, stream>>>(W2, W2bf, W2_ELEMS);
    }
    {   // 3) GEMM1 + tanh: 7 * (8192/128) * (1024/64) = 7168 blocks
        dim3 grid((unsigned)(G_ * (B_ / 128) * (H_ / 64)));
        gemm1_tanh_kernel<<<grid, TPB, 0, stream>>>(Vbf, W1bf, b1, h1bf);
    }
    {   // 4) fused gates: (8192/64) * (1024/32) = 4096 blocks
        dim3 grid((unsigned)((B_ / 64) * (H_ / 32)));
        gemm2_gate_kernel<<<grid, TPB, 0, stream>>>(h1bf, W2bf, b2, c_t, c_target, out);
    }
}
